// VanillaRNN_42451456754066
// MI455X (gfx1250) — matrix operand order repaired and checked by reading, not compiled, not run on hardware
//
#include <hip/hip_runtime.h>
#include <hip/hip_bf16.h>
#include <math.h>

typedef __attribute__((ext_vector_type(16))) __bf16 v16bf;
typedef __attribute__((ext_vector_type(8)))  __bf16 v8bf;
typedef __attribute__((ext_vector_type(8)))  float  v8f;

#define SEQ 128
#define BATCH 2048
#define HID 1024
#define NCLS 10

// gfx1250 has a hardware V_TANH_F32 (TRANS pipe, co-executes with XDL/WMMA).
// Guard the builtin name so compilation is robust across toolchains.
__device__ __forceinline__ float fast_tanh(float v) {
#if __has_builtin(__builtin_amdgcn_tanhf)
    return __builtin_amdgcn_tanhf(v);
#elif __has_builtin(__builtin_amdgcn_tanh_f32)
    return __builtin_amdgcn_tanh_f32(v);
#else
    return tanhf(v);
#endif
}

// ---------- kernel 1: W_hh (f32, [K][N] row-major) -> W_hh^T (bf16, [N][K]) ----------
__global__ __launch_bounds__(256) void k_whhT(const float* __restrict__ Whh,
                                              __bf16* __restrict__ WhhT) {
    int idx = blockIdx.x * 256 + threadIdx.x;      // 0 .. HID*HID-1
    int k = idx >> 10;                             // row of Whh (coalesced read)
    int n = idx & (HID - 1);
    WhhT[n * HID + k] = (__bf16)Whh[k * HID + n];
}

// ---------- kernel 2: t = 0 : h = tanh(x[:,0] * W_hx + b_h) ----------
__global__ __launch_bounds__(256) void k_init(const float* __restrict__ x,
                                              const float* __restrict__ Whx,
                                              const float* __restrict__ bh,
                                              __bf16* __restrict__ h) {
    int idx = blockIdx.x * 256 + threadIdx.x;      // 0 .. BATCH*HID-1
    int b = idx >> 10;
    int j = idx & (HID - 1);
    float v = fast_tanh(x[b * SEQ] * Whx[j] + bh[j]);
    h[idx] = (__bf16)v;
}

// ---------- kernel 3: one RNN step: hout = tanh(x[:,t]*W_hx + hin @ W_hh + b_h) ----------
// Block tile 64(M) x 128(N), 8 waves, each wave a 32x32 tile (2x2 WMMA bf16 tiles), K=1024.
__global__ __launch_bounds__(256) void k_step(const __bf16* __restrict__ hin,
                                              const __bf16* __restrict__ WhhT,
                                              const float*  __restrict__ x,
                                              const float*  __restrict__ Whx,
                                              const float*  __restrict__ bh,
                                              __bf16* __restrict__ hout,
                                              int t) {
    const int lane  = threadIdx.x & 31;
    const int wave  = threadIdx.x >> 5;
    const int waveM = wave & 1;          // 2 wave-rows of 32 M
    const int waveN = wave >> 1;         // 4 wave-cols of 32 N
    const int rowBase = blockIdx.x * 64  + waveM * 32;
    const int colBase = blockIdx.y * 128 + waveN * 32;
    const int l16 = lane & 15;
    const bool hi = (lane >= 16);

    // A fragment rows (matrix h, row-major [B][K])
    const int ar0 = rowBase + l16;
    const int ar1 = rowBase + 16 + l16;
    // B fragment columns == rows of WhhT ([N][K] row-major)
    const int bc0 = colBase + l16;
    const int bc1 = colBase + 16 + l16;

    // ISA fragment layout offsets (16-bit, 16x32 A / 32x16 B, wave32):
    // A: lanes 0-15 -> K {kc..kc+7, kc+16..kc+23}; lanes 16-31 -> K {kc+8..15, kc+24..31}
    // B: lanes 0-15 -> K {kc..kc+7, kc+16..kc+23}; lanes 16-31 -> K {kc+8..15, kc+24..31}
    const int aK = hi ? 8 : 0;
    const int bK = hi ? 8 : 0;

    const __bf16* pa0 = hin  + (size_t)ar0 * HID + aK;
    const __bf16* pa1 = hin  + (size_t)ar1 * HID + aK;
    const __bf16* pb0 = WhhT + (size_t)bc0 * HID + bK;
    const __bf16* pb1 = WhhT + (size_t)bc1 * HID + bK;

    v8f acc00 = {}, acc01 = {}, acc10 = {}, acc11 = {};

#pragma unroll 2
    for (int kc = 0; kc < HID; kc += 32) {
        v8bf a0lo = *(const v8bf*)(pa0 + kc);
        v8bf a0hi = *(const v8bf*)(pa0 + kc + 16);
        v8bf a1lo = *(const v8bf*)(pa1 + kc);
        v8bf a1hi = *(const v8bf*)(pa1 + kc + 16);
        v8bf b0lo = *(const v8bf*)(pb0 + kc);
        v8bf b0hi = *(const v8bf*)(pb0 + kc + 16);
        v8bf b1lo = *(const v8bf*)(pb1 + kc);
        v8bf b1hi = *(const v8bf*)(pb1 + kc + 16);

        v16bf A0 = __builtin_shufflevector(a0lo, a0hi, 0,1,2,3,4,5,6,7,8,9,10,11,12,13,14,15);
        v16bf A1 = __builtin_shufflevector(a1lo, a1hi, 0,1,2,3,4,5,6,7,8,9,10,11,12,13,14,15);
        v16bf B0 = __builtin_shufflevector(b0lo, b0hi, 0,1,2,3,4,5,6,7,8,9,10,11,12,13,14,15);
        v16bf B1 = __builtin_shufflevector(b1lo, b1hi, 0,1,2,3,4,5,6,7,8,9,10,11,12,13,14,15);

        acc00 = __builtin_amdgcn_wmma_f32_16x16x32_bf16(false, A0, false, B0, (short)0, acc00, false, false);
        acc01 = __builtin_amdgcn_wmma_f32_16x16x32_bf16(false, A0, false, B1, (short)0, acc01, false, false);
        acc10 = __builtin_amdgcn_wmma_f32_16x16x32_bf16(false, A1, false, B0, (short)0, acc10, false, false);
        acc11 = __builtin_amdgcn_wmma_f32_16x16x32_bf16(false, A1, false, B1, (short)0, acc11, false, false);
    }

    // Epilogue. C/D layout: lane l (col = colTile + l16), VGPR v -> row = rowTile + v + (hi?8:0)
    const float wx0 = Whx[bc0], wx1 = Whx[bc1];
    const float bh0 = bh[bc0],  bh1 = bh[bc1];
    const int rofs = hi ? 8 : 0;

#pragma unroll
    for (int v = 0; v < 8; ++v) {
        int r0 = rowBase + rofs + v;            // mi = 0
        int r1 = rowBase + 16 + rofs + v;       // mi = 1
        float xv0 = x[r0 * SEQ + t];
        float xv1 = x[r1 * SEQ + t];
        float c00 = acc00[v] + xv0 * wx0 + bh0;
        float c01 = acc01[v] + xv0 * wx1 + bh1;
        float c10 = acc10[v] + xv1 * wx0 + bh0;
        float c11 = acc11[v] + xv1 * wx1 + bh1;
        hout[(size_t)r0 * HID + bc0] = (__bf16)fast_tanh(c00);
        hout[(size_t)r0 * HID + bc1] = (__bf16)fast_tanh(c01);
        hout[(size_t)r1 * HID + bc0] = (__bf16)fast_tanh(c10);
        hout[(size_t)r1 * HID + bc1] = (__bf16)fast_tanh(c11);
    }
}

// ---------- kernel 4: out = h_final @ W_ph + b_p  (2048x10, f32) ----------
__global__ __launch_bounds__(256) void k_proj(const __bf16* __restrict__ h,
                                              const float* __restrict__ Wph,
                                              const float* __restrict__ bp,
                                              float* __restrict__ out) {
    int idx = blockIdx.x * 256 + threadIdx.x;
    if (idx >= BATCH * NCLS) return;
    int b = idx / NCLS;
    int c = idx - b * NCLS;
    const __bf16* hr = h + (size_t)b * HID;
    float s = bp[c];
    for (int k = 0; k < HID; k += 8) {
        v8bf hv = *(const v8bf*)(hr + k);
#pragma unroll
        for (int u = 0; u < 8; ++u)
            s += (float)hv[u] * Wph[(k + u) * NCLS + c];
    }
    out[idx] = s;
}

extern "C" void kernel_launch(void* const* d_in, const int* in_sizes, int n_in,
                              void* d_out, int out_size, void* d_ws, size_t ws_size,
                              hipStream_t stream) {
    const float* x   = (const float*)d_in[0];   // [2048,128]
    const float* Whx = (const float*)d_in[1];   // [1,1024]
    const float* Whh = (const float*)d_in[2];   // [1024,1024]
    const float* Wph = (const float*)d_in[3];   // [1024,10]
    const float* bh  = (const float*)d_in[4];   // [1,1024]
    const float* bp  = (const float*)d_in[5];   // [1,10]
    float* out = (float*)d_out;

    char* ws = (char*)d_ws;
    __bf16* WhhT = (__bf16*)ws;                                       // 2 MB
    __bf16* hA   = (__bf16*)(ws + (size_t)HID * HID * 2);             // 4 MB
    __bf16* hB   = (__bf16*)(ws + (size_t)HID * HID * 2
                                + (size_t)BATCH * HID * 2);           // 4 MB

    k_whhT<<<(HID * HID) / 256, 256, 0, stream>>>(Whh, WhhT);
    k_init<<<(BATCH * HID) / 256, 256, 0, stream>>>(x, Whx, bh, hA);

    __bf16* cur = hA;
    __bf16* nxt = hB;
    for (int t = 1; t < SEQ; ++t) {
        k_step<<<dim3(BATCH / 64, HID / 128), 256, 0, stream>>>(cur, WhhT, x, Whx, bh, nxt, t);
        __bf16* tmp = cur; cur = nxt; nxt = tmp;
    }

    k_proj<<<(BATCH * NCLS + 255) / 256, 256, 0, stream>>>(cur, Wph, bp, out);
}
